// Memory_efficient_network_40535901340193
// MI455X (gfx1250) — matrix operand order repaired and checked by reading, not compiled, not run on hardware
//
#include <hip/hip_runtime.h>
#include <hip/hip_bf16.h>

typedef __attribute__((ext_vector_type(16))) __bf16         bf16x16;
typedef __attribute__((ext_vector_type(8)))  float          f32x8;
typedef __attribute__((ext_vector_type(4)))  float          f32x4;
typedef __attribute__((ext_vector_type(8)))  unsigned short u16x8;

struct Frag { u16x8 lo, hi; };   // 32B, matches bf16x16

// ---------- helpers ----------
__device__ __forceinline__ unsigned short f2bf(float f) {
    unsigned int u = __float_as_uint(f);
    unsigned int r = u + 0x7FFFu + ((u >> 16) & 1u);   // round-to-nearest-even
    return (unsigned short)(r >> 16);
}
__device__ __forceinline__ float bf2f(unsigned short u) {
    return __uint_as_float((unsigned int)u << 16);
}
__device__ __forceinline__ float lrelu(float v) { return v > 0.f ? v : 0.01f * v; }
__device__ __forceinline__ unsigned short to_bf(float f)          { return f2bf(f); }
__device__ __forceinline__ unsigned short to_bf(unsigned short u) { return u; }

// order-independent float atomic max (signed-max / unsigned-min bit trick)
__device__ __forceinline__ void atomicMaxF32(float* addr, float val) {
    if (val >= 0.f) atomicMax((int*)addr, __float_as_int(val));
    else            atomicMin((unsigned int*)addr, (unsigned int)__float_as_int(val));
}

// ---------- fill with -inf (atomic-max identity) ----------
__global__ void fill_neginf(float* __restrict__ p, int n) {
    int i = blockIdx.x * blockDim.x + threadIdx.x;
    if (i < n) p[i] = __int_as_float(0xFF800000);
}

// ---------- weight prep: W[64, ld] fp32 -> Wt[n(64)][k(64)] bf16, zero padded ----------
__global__ void prep_weights(const float* __restrict__ W, unsigned short* __restrict__ Wt,
                             int CIN, int ld) {
    int e = blockIdx.x * blockDim.x + threadIdx.x;
    if (e >= 64 * 64) return;
    int n = e >> 6, k = e & 63;
    float v = (k < CIN) ? W[n * ld + k] : 0.f;
    Wt[e] = f2bf(v);
}

// ---------- max over middle axis: x[outer][R][inner] -> y[outer][inner] (inner = 1<<ish) ----------
__global__ void max_over_axis(const float* __restrict__ x, float* __restrict__ y,
                              int R, int ish, int total) {
    int idx = blockIdx.x * blockDim.x + threadIdx.x;
    if (idx >= total) return;
    int o = idx >> ish, i = idx & ((1 << ish) - 1);
    const float* p = x + (((size_t)o * R) << ish) + i;
    float m = -3.4e38f;
    for (int r = 0; r < R; ++r) m = fmaxf(m, p[(size_t)r << ish]);
    y[idx] = m;
}
__global__ void max_over_axis_bf(const unsigned short* __restrict__ x, float* __restrict__ y,
                                 int R, int ish, int total) {
    int idx = blockIdx.x * blockDim.x + threadIdx.x;
    if (idx >= total) return;
    int o = idx >> ish, i = idx & ((1 << ish) - 1);
    const unsigned short* p = x + (((size_t)o * R) << ish) + i;
    float m = -3.4e38f;
    for (int r = 0; r < R; ++r) m = fmaxf(m, bf2f(p[(size_t)r << ish]));
    y[idx] = m;
}

// ---------- low-rank correction term: y[b][64][L] = W[:, colOff:colOff+CIN] @ xm[b][CIN][L] ----------
__global__ void term_gemm(const float* __restrict__ W, int colOff, int ld, int CIN,
                          const float* __restrict__ xm, float* __restrict__ y,
                          int lsh, int total) {
    int idx = blockIdx.x * blockDim.x + threadIdx.x;
    if (idx >= total) return;
    int i = idx & ((1 << lsh) - 1);
    int bn = idx >> lsh;
    int n = bn & 63, b = bn >> 6;
    float acc = 0.f;
    for (int ch = 0; ch < CIN; ++ch)
        acc += W[n * ld + colOff + ch] * xm[(((size_t)b * CIN + ch) << lsh) + i];
    y[idx] = acc;
}

// ---------- main WMMA pconv + rank-correction + bias + leaky-relu ----------
// X: [B][CIN][S], Wt: bf16 [n(64)][k(64)], Y: bf16 [B][64][S] (if STORE_Y)
// rmax: [B][64][D1][2] fused atomic row-max of outputs (if FUSE_RM)
template<int KS, int CIN, int D1, int D2X2, int SH2, bool STORE_Y, bool FUSE_RM, typename TIN>
__global__ void __launch_bounds__(128)
pconv_wmma(const TIN* __restrict__ X, const unsigned short* __restrict__ Wt,
           const float* __restrict__ termA, const float* __restrict__ termB,
           const float* __restrict__ bias, unsigned short* __restrict__ Y,
           float* __restrict__ rmax) {
    constexpr int PITCH = 72;                 // shorts per row: 144 B, 16B aligned
    constexpr int S = D1 * D2X2;
    __shared__ __align__(16) unsigned short lA[64 * PITCH];  // [m][k]
    __shared__ __align__(16) unsigned short lB[64 * PITCH];  // [n][k]
    const int b = blockIdx.y;
    const int spBase = blockIdx.x * 64;
    const int tid = threadIdx.x;

    // ---- stage X tile -> bf16 [m][k]; load phase batched, then convert/pack ----
    {
        const int mfill = tid & 63;
        const int kg0 = (tid >> 6) * 8;       // 0 or 8
        const size_t xbase = (size_t)b * CIN * S + spBase + mfill;
        TIN vals[KS * 2][8];
#pragma unroll
        for (int h = 0; h < KS * 2; ++h) {
#pragma unroll
            for (int j = 0; j < 8; ++j) {
                int k = kg0 + h * 16 + j;                  // compile-time predicate
                vals[h][j] = (k < CIN) ? X[xbase + (size_t)k * S] : (TIN)0;
            }
        }
#pragma unroll
        for (int h = 0; h < KS * 2; ++h) {
            u16x8 pack;
#pragma unroll
            for (int j = 0; j < 8; ++j) pack[j] = to_bf(vals[h][j]);
            *(u16x8*)&lA[mfill * PITCH + kg0 + h * 16] = pack;
        }
        if (blockIdx.x + 1 < gridDim.x)                    // global_prefetch_b8
            __builtin_prefetch(&X[xbase + 64], 0, 1);
    }
    // ---- stage weights: b128 copies ----
    for (int idx = tid; idx < 512; idx += 128) {
        int n = idx >> 3, k0 = (idx & 7) << 3;
        *(u16x8*)&lB[n * PITCH + k0] = *(const u16x8*)&Wt[n * 64 + k0];
    }
    __syncthreads();

    const int lane = tid & 31;
    const int wv   = tid >> 5;
    const int hi   = lane >> 4;
    const int mloc = wv * 16 + (lane & 15);

    f32x8 z = {0.f, 0.f, 0.f, 0.f, 0.f, 0.f, 0.f, 0.f};
    f32x8 acc[4] = {z, z, z, z};

#pragma unroll
    for (int kt = 0; kt < KS; ++kt) {
        const int kA0 = kt * 32 + hi * 8;     // ISA 16-bit 16x32 A layout: two 8-short runs
        const int kB0 = kt * 32 + hi * 8;
        Frag fa;
        fa.lo = *(const u16x8*)&lA[mloc * PITCH + kA0];
        fa.hi = *(const u16x8*)&lA[mloc * PITCH + kA0 + 16];
        bf16x16 afrag = __builtin_bit_cast(bf16x16, fa);
#pragma unroll
        for (int nt = 0; nt < 4; ++nt) {
            const int nn = nt * 16 + (lane & 15);
            Frag fb;
            fb.lo = *(const u16x8*)&lB[nn * PITCH + kB0];
            fb.hi = *(const u16x8*)&lB[nn * PITCH + kB0 + 16];
            bf16x16 bfrag = __builtin_bit_cast(bf16x16, fb);
            acc[nt] = __builtin_amdgcn_wmma_f32_16x16x32_bf16(
                false, afrag, false, bfrag, (short)0, acc[nt], false, false);
        }
    }

    // ---- epilogue: D layout m = r + hi*8, n = lane&15 ----
    const int sp0  = spBase + wv * 16 + hi * 8;
    const int d1   = spBase >> SH2;
    const int rem0 = sp0 & (D2X2 - 1);
#pragma unroll
    for (int nt = 0; nt < 4; ++nt) {
        const int n = nt * 16 + (lane & 15);
        const float bn = bias[n];
        const size_t baseOut = ((size_t)b * 64 + n) * S;
        const size_t baseA   = (((size_t)b * 64 + n) * D1 + d1) * 2;
        const size_t baseB   = ((size_t)b * 64 + n) * (size_t)D2X2 + rem0;
        const float tA0 = termA[baseA];
        const float tA1 = termA[baseA + 1];
        f32x4 tb0 = *(const f32x4*)&termB[baseB];
        f32x4 tb1 = *(const f32x4*)&termB[baseB + 4];
        float v[8];
#pragma unroll
        for (int r = 0; r < 4; ++r) {
            float e = (r & 1) ? tA1 : tA0;
            v[r]     = lrelu(acc[nt][r]     + e + tb0[r] + bn);
            v[r + 4] = lrelu(acc[nt][r + 4] + e + tb1[r] + bn);
        }
        if constexpr (STORE_Y) {
            u16x8 ob;
#pragma unroll
            for (int r = 0; r < 8; ++r) ob[r] = f2bf(v[r]);
            *(u16x8*)&Y[baseOut + sp0] = ob;
        }
        if constexpr (FUSE_RM) {
            // reduce over d2 within the tile: 8 values -> even/odd parity maxes
            float me = fmaxf(fmaxf(v[0], v[2]), fmaxf(v[4], v[6]));
            float mo = fmaxf(fmaxf(v[1], v[3]), fmaxf(v[5], v[7]));
            me = fmaxf(me, __shfl_xor(me, 16, 32));   // merge lane L <-> L+16 (same n)
            mo = fmaxf(mo, __shfl_xor(mo, 16, 32));
            if (hi == 0) {
                atomicMaxF32(&rmax[baseA],     me);
                atomicMaxF32(&rmax[baseA + 1], mo);
            }
        }
    }
}

// ---------- tiny pconv + lrelu for p / g / one chains (L = 1<<lsh) ----------
__global__ void small_pconv(const float* __restrict__ W, const float* __restrict__ bias,
                            int CIN, const float* __restrict__ x, float* __restrict__ y,
                            int lsh, int total) {
    int idx = blockIdx.x * blockDim.x + threadIdx.x;
    if (idx >= total) return;
    int i = idx & ((1 << lsh) - 1);
    int bn = idx >> lsh;
    int n = bn & 63, b = bn >> 6;
    float acc = bias[n];
    for (int ch = 0; ch < CIN; ++ch)
        acc += W[n * CIN + ch] * x[(((size_t)b * CIN + ch) << lsh) + i];
    y[idx] = lrelu(acc);
}

// ---------- heads ----------
__global__ void head2(const float* __restrict__ W, const float* __restrict__ bias,
                      const float* __restrict__ pooled, const float* __restrict__ m,
                      float* __restrict__ out, int lsh, int total) {
    int idx = blockIdx.x * blockDim.x + threadIdx.x;
    if (idx >= total) return;
    int i = idx & ((1 << lsh) - 1), b = idx >> lsh;
    int s = i & 1;
    float acc = bias[0];
    for (int n = 0; n < 64; ++n)
        acc += W[n] * pooled[(((size_t)b * 64 + n) << lsh) + i] + W[64 + n] * m[(b * 64 + n) * 2 + s];
    out[idx] = acc;
}

__global__ void head1(const float* __restrict__ W, const float* __restrict__ bias,
                      const float* __restrict__ x, float* __restrict__ out,
                      int lsh, int total) {
    int idx = blockIdx.x * blockDim.x + threadIdx.x;
    if (idx >= total) return;
    int i = idx & ((1 << lsh) - 1), b = idx >> lsh;
    float acc = bias[0];
    for (int n = 0; n < 64; ++n)
        acc += W[n] * x[(((size_t)b * 64 + n) << lsh) + i];
    out[idx] = acc;
}

// ---------- final scalar value ----------
__global__ void value_kernel(const float* vgxc, int n1, const float* vpxg, int n2,
                             const float* vp, int n3, const float* vg, int n4,
                             const float* v1, int n5, float* out) {
    __shared__ float red[256];
    int tid = threadIdx.x;
    float s = 0.f;
    for (int i = tid; i < n1; i += 256) s += 0.5f * vgxc[i];
    for (int i = tid; i < n2; i += 256) s += 0.5f * vpxg[i];
    for (int i = tid; i < n3; i += 256) s += vp[i];
    for (int i = tid; i < n4; i += 256) s += vg[i];
    for (int i = tid; i < n5; i += 256) s += v1[i];
    red[tid] = s;
    __syncthreads();
    for (int off = 128; off > 0; off >>= 1) {
        if (tid < off) red[tid] += red[tid + off];
        __syncthreads();
    }
    if (tid == 0) out[0] = red[0];
}

// ======================================================================
extern "C" void kernel_launch(void* const* d_in, const int* in_sizes, int n_in,
                              void* d_out, int out_size, void* d_ws, size_t ws_size,
                              hipStream_t stream) {
    (void)in_sizes; (void)n_in; (void)out_size; (void)ws_size;
    const int B = 2, G = 256, C = 512, P = 128;
    const int Sg = G * C * 2;   // 262144
    const int Sp = P * G * 2;   // 65536

    const float* xGXC = (const float*)d_in[0];
    const float* xPXG = (const float*)d_in[1];
    const float* xP   = (const float*)d_in[2];
    const float* xG   = (const float*)d_in[3];
    float* out = (float*)d_out;

    // workspace carve-out (float units; bf16 buffers use half)
    float* ws = (float*)d_ws;
    size_t off = 0;
    auto alloc = [&](size_t n) { float* p = ws + off; off += n; return p; };
    unsigned short* gxcA = (unsigned short*)alloc((size_t)B * 64 * Sg / 2);  // bf16
    unsigned short* gxcB = (unsigned short*)alloc((size_t)B * 64 * Sg / 2);  // bf16
    unsigned short* pxgA = (unsigned short*)alloc((size_t)B * 64 * Sp / 2);  // bf16
    float* rmG1 = alloc(65536);   // fused rowmax: maxC input for L1 gxc [B,64,G,2]
    float* rmP1 = alloc(32768);   // fused rowmax: maxC input for L1 pxg [B,64,P,2]
    float* plPf = alloc(32768);   // fused pooled_pxg [B,64,P,2]
    float* maxCb = alloc(65536);
    float* maxGb = alloc(131072);
    float* tA = alloc(65536);
    float* tB = alloc(131072);
    unsigned short* wt = (unsigned short*)alloc(4 * 2048);  // 4 x 4096 bf16
    unsigned short* wt0 = wt, *wt1 = wt + 4096, *wt2 = wt + 8192, *wt3 = wt + 12288;
    float* pA = alloc(16384);  float* pB = alloc(16384);
    float* gA = alloc(32768);  float* gB = alloc(32768);
    float* oneB = alloc(32768);
    float* plG = alloc(131072);  // pooled_gxc [B,64,C,2] fp32
    float* mG = alloc(256); float* mP = alloc(256);
    float* vGxc = alloc(2048); float* vPxg = alloc(512);
    float* vP = alloc(256); float* vG = alloc(512); float* v1 = alloc(512);

    auto gd = [](int n) { return dim3((unsigned)((n + 255) / 256)); };

    // ---- init fused-max buffers (rmG1, rmP1, plPf are contiguous: 131072 floats) ----
    fill_neginf<<<gd(131072), 256, 0, stream>>>(rmG1, 131072);

    // ---- weight prep (bf16, n-major K-contiguous, zero padded) ----
    prep_weights<<<16, 256, 0, stream>>>((const float*)d_in[5],  wt0, 16, 48);
    prep_weights<<<16, 256, 0, stream>>>((const float*)d_in[7],  wt1, 16, 48);
    prep_weights<<<16, 256, 0, stream>>>((const float*)d_in[15], wt2, 64, 192);
    prep_weights<<<16, 256, 0, stream>>>((const float*)d_in[17], wt3, 64, 192);

    // ================= layer 0 (fp32 inputs -> bf16 activations, fused next-layer rowmax) ====
    // gxc: CIN=16, D1=G, D2x2=1024
    max_over_axis<<<gd(B*16*G*2), 256, 0, stream>>>(xGXC, maxCb, C, 1, B*16*G*2);
    max_over_axis<<<gd(B*16*C*2), 256, 0, stream>>>(xGXC, maxGb, G, 10, B*16*C*2);
    term_gemm<<<gd(B*64*G*2), 256, 0, stream>>>((const float*)d_in[5], 16, 48, 16, maxCb, tA, 9,  B*64*G*2);
    term_gemm<<<gd(B*64*C*2), 256, 0, stream>>>((const float*)d_in[5], 32, 48, 16, maxGb, tB, 10, B*64*C*2);
    pconv_wmma<1, 16, 256, 1024, 10, true, true, float><<<dim3(Sg/64, B), 128, 0, stream>>>(
        xGXC, wt0, tA, tB, (const float*)d_in[6], gxcA, rmG1);
    // pxg: CIN=16, D1=P, D2x2=512
    max_over_axis<<<gd(B*16*P*2), 256, 0, stream>>>(xPXG, maxCb, G, 1, B*16*P*2);
    max_over_axis<<<gd(B*16*G*2), 256, 0, stream>>>(xPXG, maxGb, P, 9, B*16*G*2);
    term_gemm<<<gd(B*64*P*2), 256, 0, stream>>>((const float*)d_in[7], 16, 48, 16, maxCb, tA, 8, B*64*P*2);
    term_gemm<<<gd(B*64*G*2), 256, 0, stream>>>((const float*)d_in[7], 32, 48, 16, maxGb, tB, 9, B*64*G*2);
    pconv_wmma<1, 16, 128, 512, 9, true, true, float><<<dim3(Sp/64, B), 128, 0, stream>>>(
        xPXG, wt1, tA, tB, (const float*)d_in[8], pxgA, rmP1);
    // p / g / one chains (fp32 throughout, tiny)
    small_pconv<<<gd(B*64*P), 256, 0, stream>>>((const float*)d_in[9],  (const float*)d_in[10], 16, xP, pA, 7, B*64*P);
    small_pconv<<<gd(B*64*G), 256, 0, stream>>>((const float*)d_in[11], (const float*)d_in[12], 16, xG, gA, 8, B*64*G);
    small_pconv<<<gd(B*64*G), 256, 0, stream>>>((const float*)d_in[13], (const float*)d_in[14], 64, gA, oneB, 8, B*64*G);

    // ================= layer 1 (bf16 in; rowmax already fused) =================
    // gxc: CIN=64  (termA from fused rmG1; colmax pass still needed)
    max_over_axis_bf<<<gd(B*64*C*2), 256, 0, stream>>>(gxcA, maxGb, G, 10, B*64*C*2);
    term_gemm<<<gd(B*64*G*2), 256, 0, stream>>>((const float*)d_in[15], 64, 192, 64, rmG1, tA, 9,  B*64*G*2);
    term_gemm<<<gd(B*64*C*2), 256, 0, stream>>>((const float*)d_in[15], 128, 192, 64, maxGb, tB, 10, B*64*C*2);
    pconv_wmma<2, 64, 256, 1024, 10, true, false, unsigned short><<<dim3(Sg/64, B), 128, 0, stream>>>(
        gxcA, wt2, tA, tB, (const float*)d_in[16], gxcB, nullptr);
    // pxg: CIN=64  (termA from fused rmP1; output never stored, pooled via fused atomics)
    max_over_axis_bf<<<gd(B*64*G*2), 256, 0, stream>>>(pxgA, maxGb, P, 9, B*64*G*2);
    term_gemm<<<gd(B*64*P*2), 256, 0, stream>>>((const float*)d_in[17], 64, 192, 64, rmP1, tA, 8, B*64*P*2);
    term_gemm<<<gd(B*64*G*2), 256, 0, stream>>>((const float*)d_in[17], 128, 192, 64, maxGb, tB, 9, B*64*G*2);
    pconv_wmma<2, 64, 128, 512, 9, false, true, unsigned short><<<dim3(Sp/64, B), 128, 0, stream>>>(
        pxgA, wt3, tA, tB, (const float*)d_in[18], nullptr, plPf);
    // p / g / one chains
    small_pconv<<<gd(B*64*P), 256, 0, stream>>>((const float*)d_in[19], (const float*)d_in[20], 64, pA, pB, 7, B*64*P);
    small_pconv<<<gd(B*64*G), 256, 0, stream>>>((const float*)d_in[21], (const float*)d_in[22], 64, gA, gB, 8, B*64*G);
    small_pconv<<<gd(B*64*G), 256, 0, stream>>>((const float*)d_in[23], (const float*)d_in[24], 64, gB, oneB, 8, B*64*G);

    // ================= pooling =================
    max_over_axis_bf<<<gd(B*64*C*2), 256, 0, stream>>>(gxcB, plG, G, 10, B*64*C*2);  // pooled_gxc
    max_over_axis<<<gd(B*64*2), 256, 0, stream>>>(plG,  mG, C, 1, B*64*2);
    max_over_axis<<<gd(B*64*2), 256, 0, stream>>>(plPf, mP, P, 1, B*64*2);

    // ================= heads =================
    head2<<<gd(B*C*2), 256, 0, stream>>>((const float*)d_in[25], (const float*)d_in[26], plG,  mG, out + 0,    10, B*C*2);
    head2<<<gd(B*P*2), 256, 0, stream>>>((const float*)d_in[27], (const float*)d_in[28], plPf, mP, out + 2048, 8,  B*P*2);
    head1<<<gd(B*P),   256, 0, stream>>>((const float*)d_in[29], (const float*)d_in[30], pB,   out + 2560, 7, B*P);
    head1<<<gd(B*G),   256, 0, stream>>>((const float*)d_in[31], (const float*)d_in[32], gB,   out + 2816, 8, B*G);
    head1<<<gd(B*G),   256, 0, stream>>>((const float*)d_in[33], (const float*)d_in[34], oneB, out + 3328, 8, B*G);
    head2<<<gd(B*C*2), 256, 0, stream>>>((const float*)d_in[35], (const float*)d_in[36], plG,  mG, vGxc, 10, B*C*2);
    head2<<<gd(B*P*2), 256, 0, stream>>>((const float*)d_in[37], (const float*)d_in[38], plPf, mP, vPxg, 8,  B*P*2);
    head1<<<gd(B*P),   256, 0, stream>>>((const float*)d_in[39], (const float*)d_in[40], pB,   vP, 7, B*P);
    head1<<<gd(B*G),   256, 0, stream>>>((const float*)d_in[41], (const float*)d_in[42], gB,   vG, 8, B*G);
    head1<<<gd(B*G),   256, 0, stream>>>((const float*)d_in[43], (const float*)d_in[44], oneB, v1, 8, B*G);

    value_kernel<<<1, 256, 0, stream>>>(vGxc, 2048, vPxg, 512, vP, 256, vG, 512, v1, 512, out + 3840);
}
